// EncoderBlock_73650099192433
// MI455X (gfx1250) — compile-verified, matrix operand order checked by reading, not run on hardware
//
#include <hip/hip_runtime.h>
#include <math.h>
#include <stdint.h>

typedef __attribute__((ext_vector_type(16))) _Float16 v16h;
typedef __attribute__((ext_vector_type(8)))  _Float16 h8;
typedef __attribute__((ext_vector_type(8)))  float    v8f;

#define D_MODEL 768
#define N_HEADS 12
#define D_FF    3072
#define BATCH   2
#define SEQ     2048
#define TOKENS  (BATCH*SEQ)   /* 4096 */
#define DK      64

// ---------------------------------------------------------------------------
// Fragment load (works for global or LDS pointers): row-major f16 source,
// lane l holds row (l&15); element e: K = 8*(l>>4)+e (e<8), 16+8*(l>>4)+(e-8).
// Caller passes p = base + row*ld + k0 + 8*(lane>>4).
// ---------------------------------------------------------------------------
__device__ __forceinline__ v16h ldfrag(const _Float16* p) {
  v16h f;
  ((h8*)&f)[0] = *(const h8*)(p);
  ((h8*)&f)[1] = *(const h8*)(p + 16);
  return f;
}

__device__ __forceinline__ v8f wmma_f16(v16h a, v16h b, v8f c) {
  return __builtin_amdgcn_wmma_f32_16x16x32_f16(false, a, false, b,
                                                (short)0, c, false, false);
}

// ---------------------------------------------------------------------------
// CDNA5 async global->LDS copy (16B per lane), tracked by ASYNCcnt.
// VDST = LDS byte address (generic LDS addr low 32 bits == LDS offset),
// VADDR = 64-bit global address.
// ---------------------------------------------------------------------------
__device__ __forceinline__ void async_cp16(_Float16* lds, const _Float16* g) {
  unsigned l = (unsigned)(uintptr_t)lds;   // low 32 bits of generic LDS addr
  asm volatile("global_load_async_to_lds_b128 %0, %1, off"
               :: "v"(l), "v"(g) : "memory");
}
#define WAIT_ASYNC(N) asm volatile("s_wait_asynccnt " #N ::: "memory")

// ---------------------------------------------------------------------------
// Weight prep: w [K,N] fp32 row-major  ->  wt [N,K] f16 row-major
// ---------------------------------------------------------------------------
__global__ __launch_bounds__(256) void transpose_cast_kernel(
    const float* __restrict__ w, _Float16* __restrict__ wt, int K, int N)
{
  int idx = blockIdx.x * 256 + threadIdx.x;   // linear over wt
  int n = idx / K, k = idx - n * K;
  wt[idx] = (_Float16)w[(size_t)k * N + n];
}

// ---------------------------------------------------------------------------
// LayerNorm (torch semantics: unbiased std, scalar alpha/beta, eps on std)
// ---------------------------------------------------------------------------
__global__ __launch_bounds__(256) void ln_kernel(
    const float* __restrict__ x, _Float16* __restrict__ out,
    const float* __restrict__ ap, const float* __restrict__ bp)
{
  __shared__ float red[256];
  const int row = blockIdx.x, t = threadIdx.x;
  const float* xr = x + (size_t)row * D_MODEL;
  float v0 = xr[t], v1 = xr[t + 256], v2 = xr[t + 512];

  red[t] = v0 + v1 + v2;
  __syncthreads();
  for (int off = 128; off > 0; off >>= 1) {
    if (t < off) red[t] += red[t + off];
    __syncthreads();
  }
  float mean = red[0] * (1.0f / 768.0f);
  __syncthreads();

  float d0 = v0 - mean, d1 = v1 - mean, d2 = v2 - mean;
  red[t] = d0 * d0 + d1 * d1 + d2 * d2;
  __syncthreads();
  for (int off = 128; off > 0; off >>= 1) {
    if (t < off) red[t] += red[t + off];
    __syncthreads();
  }
  float stdv = sqrtf(red[0] * (1.0f / 767.0f));   // Bessel (ddof=1)
  float s = ap[0] / (stdv + 1e-6f);
  float beta = bp[0];

  _Float16* orow = out + (size_t)row * D_MODEL;
  orow[t]       = (_Float16)(d0 * s + beta);
  orow[t + 256] = (_Float16)(d1 * s + beta);
  orow[t + 512] = (_Float16)(d2 * s + beta);
}

// ---------------------------------------------------------------------------
// WMMA GEMM with async-LDS double-buffered staging.
// Block = 256 thr (8 waves), block tile 128x128, wave tile 32x64 (2x4 WMMA).
// LDS tiles padded to 40-half rows (80B = 20 banks -> conflict-free b128).
// ---------------------------------------------------------------------------
#define LDA_PAD 40

template<bool RELU, bool HASRES, bool OUTF16E, bool OUTF32E, bool OUTT>
__global__ __launch_bounds__(256) void gemm_kernel(
    const _Float16* __restrict__ A, const _Float16* __restrict__ Wt,
    const float* __restrict__ bias, const float* __restrict__ res,
    float* __restrict__ out32, _Float16* __restrict__ out16,
    _Float16* __restrict__ outT, int M, int N, int K)
{
  __shared__ _Float16 As[2][128 * LDA_PAD];
  __shared__ _Float16 Bs[2][128 * LDA_PAD];

  const int lane = threadIdx.x & 31;
  const int wave = threadIdx.x >> 5;
  const int half = lane >> 4;
  const int lrow = lane & 15;
  const int bm = blockIdx.y * 128, bn = blockIdx.x * 128;
  const int wrow = (wave >> 1) * 32;   // wave M offset in block tile
  const int wcol = (wave & 1) * 64;    // wave N offset in block tile

  v8f acc[2][4] = {};

  // issue async copies of one 128x32 K-slab of A and B into buffer `buf`
  auto issue = [&](int buf, int kk) {
    #pragma unroll
    for (int i = 0; i < 2; ++i) {
      int c = threadIdx.x + i * 256;          // 512 chunks of 16B each
      int row = c >> 2, kc = (c & 3) * 8;
      async_cp16(&As[buf][row * LDA_PAD + kc],
                 A  + (size_t)(bm + row) * K + kk + kc);
      async_cp16(&Bs[buf][row * LDA_PAD + kc],
                 Wt + (size_t)(bn + row) * K + kk + kc);
    }
  };

  issue(0, 0);
  const int nsteps = K / 32;
  for (int s = 0; s < nsteps; ++s) {
    const int buf = s & 1;
    __syncthreads();                       // all reads of buf^1 finished
    if (s + 1 < nsteps) {
      issue(buf ^ 1, (s + 1) * 32);
      WAIT_ASYNC(0x4);                     // my 4 copies into `buf` done
    } else {
      WAIT_ASYNC(0x0);
    }
    __syncthreads();                       // everyone's copies into `buf` done

    v16h af[2], bf[4];
    #pragma unroll
    for (int mt = 0; mt < 2; ++mt)
      af[mt] = ldfrag(&As[buf][(wrow + mt * 16 + lrow) * LDA_PAD + 8 * half]);
    #pragma unroll
    for (int nt = 0; nt < 4; ++nt)
      bf[nt] = ldfrag(&Bs[buf][(wcol + nt * 16 + lrow) * LDA_PAD + 8 * half]);
    #pragma unroll
    for (int mt = 0; mt < 2; ++mt)
      #pragma unroll
      for (int nt = 0; nt < 4; ++nt)
        acc[mt][nt] = wmma_f16(af[mt], bf[nt], acc[mt][nt]);
  }

  // Epilogue. C layout: lane col N = lrow, VGPR v row M = v + 8*half.
  #pragma unroll
  for (int nt = 0; nt < 4; ++nt) {
    const int gn = bn + wcol + nt * 16 + lrow;
    const float bv = bias[gn];
    #pragma unroll
    for (int mt = 0; mt < 2; ++mt) {
      const int gm0 = bm + wrow + mt * 16 + 8 * half;
      if (OUTT) {
        // Vt layout [B, N(=H*dk), S]: token index is the fast dim.
        const int bb = gm0 / SEQ, s0 = gm0 - bb * SEQ;
        h8 pk;
        #pragma unroll
        for (int v = 0; v < 8; ++v) pk[v] = (_Float16)(acc[mt][nt][v] + bv);
        *(h8*)(outT + ((size_t)(bb * N + gn)) * SEQ + s0) = pk;
      } else {
        #pragma unroll
        for (int v = 0; v < 8; ++v) {
          float val = acc[mt][nt][v] + bv;
          if (HASRES) val += res[(size_t)(gm0 + v) * N + gn];
          if (RELU)   val = fmaxf(val, 0.0f);
          if (OUTF32E) out32[(size_t)(gm0 + v) * N + gn] = val;
          if (OUTF16E) out16[(size_t)(gm0 + v) * N + gn] = (_Float16)val;
        }
      }
    }
  }
}

// ---------------------------------------------------------------------------
// Flash attention, K/V shared through LDS by async copies.
// Block = 4 waves = 4 q-tiles of the SAME (b,h); 32 keys per iteration.
// S^T = K_tile(A) @ Q^T(B)  ->  C layout: lane = query col, VGPR = key row,
// which is exactly the B-frag layout of P^T after a per-lane f16 pack, so
// O^T = V^T(A, from Vt) @ P^T(B) needs no cross-lane movement.
// ---------------------------------------------------------------------------
#define LDK_PAD 72   /* 64-half K rows padded: 144B = 36 banks, conflict-free */
#define LDV_PAD 40   /* 32-half V rows padded: 80B  = 20 banks, conflict-free */

__global__ __launch_bounds__(128) void attn_kernel(
    const _Float16* __restrict__ q16, const _Float16* __restrict__ k16,
    const _Float16* __restrict__ vt,  const int* __restrict__ mask,
    _Float16* __restrict__ ctx)
{
  __shared__ _Float16 Ks[2][32 * LDK_PAD];
  __shared__ _Float16 Vs[2][64 * LDV_PAD];

  const int lane = threadIdx.x & 31;
  const int wave = threadIdx.x >> 5;
  const int half = lane >> 4;
  const int lq   = lane & 15;
  const int bh   = blockIdx.x / 32;          // 0..23 : same (b,h) per block
  const int h    = bh % N_HEADS;
  const int b    = bh / N_HEADS;
  const int qt   = (blockIdx.x % 32) * 4 + wave;

  // Q^T B-frags (query rows contiguous), pre-scaled by 1/sqrt(dk)=0.125
  const _Float16* qrow =
      q16 + (size_t)(b * SEQ + qt * 16 + lq) * D_MODEL + h * DK + half * 8;
  v16h bq0 = ldfrag(qrow);
  v16h bq1 = ldfrag(qrow + 32);
  #pragma unroll
  for (int i = 0; i < 16; ++i) {
    bq0[i] = bq0[i] * (_Float16)0.125f;
    bq1[i] = bq1[i] * (_Float16)0.125f;
  }

  const _Float16* kg = k16 + (size_t)(b * SEQ) * D_MODEL + h * DK;
  const _Float16* vg = vt + (size_t)(b * N_HEADS + h) * DK * SEQ;
  const int* mrow = mask + b * SEQ;

  // stage 32 keys of K (32x64) and V^T (64x32) into buffer `buf`
  auto issue = [&](int buf, int k0) {
    #pragma unroll
    for (int i = 0; i < 2; ++i) {
      int c = threadIdx.x + i * 128;                 // 256 chunks each
      int kr = c >> 3, kc = (c & 7) * 8;
      async_cp16(&Ks[buf][kr * LDK_PAD + kc],
                 kg + (size_t)(k0 + kr) * D_MODEL + kc);
      int vr = c >> 2, vc = (c & 3) * 8;
      async_cp16(&Vs[buf][vr * LDV_PAD + vc],
                 vg + (size_t)vr * SEQ + k0 + vc);
    }
  };

  v8f o[4] = {};
  float mrun = -__builtin_inff(), lrun = 0.0f;

  issue(0, 0);
  const int nsteps = SEQ / 32;
  for (int s = 0; s < nsteps; ++s) {
    const int buf = s & 1;
    const int k0 = s * 32;
    __syncthreads();
    if (s + 1 < nsteps) {
      issue(buf ^ 1, k0 + 32);
      WAIT_ASYNC(0x4);
    } else {
      WAIT_ASYNC(0x0);
    }
    __syncthreads();

    v16h ka0 = ldfrag(&Ks[buf][(lq)      * LDK_PAD + 8 * half]);
    v16h ka1 = ldfrag(&Ks[buf][(lq)      * LDK_PAD + 8 * half + 32]);
    v16h kc0 = ldfrag(&Ks[buf][(16 + lq) * LDK_PAD + 8 * half]);
    v16h kc1 = ldfrag(&Ks[buf][(16 + lq) * LDK_PAD + 8 * half + 32]);
    v8f s0 = {};
    s0 = wmma_f16(ka0, bq0, s0);
    s0 = wmma_f16(ka1, bq1, s0);
    v8f s1 = {};
    s1 = wmma_f16(kc0, bq0, s1);
    s1 = wmma_f16(kc1, bq1, s1);

    // Key mask bits: lane j (0..15) loads mask for key j; ballot bit kl tests.
    unsigned bal0 = __builtin_amdgcn_ballot_w32(mrow[k0 + lq] != 0);
    unsigned bal1 = __builtin_amdgcn_ballot_w32(mrow[k0 + 16 + lq] != 0);

    float p[16];
    float tmax = -__builtin_inff();
    #pragma unroll
    for (int v = 0; v < 8; ++v) {
      const int kl = v + 8 * half;
      float a = ((bal0 >> kl) & 1u) ? s0[v] : -1e9f;
      float c = ((bal1 >> kl) & 1u) ? s1[v] : -1e9f;
      p[v] = a; p[v + 8] = c;
      tmax = fmaxf(tmax, fmaxf(a, c));
    }
    tmax = fmaxf(tmax, __shfl_xor(tmax, 16, 32));
    const float newm = fmaxf(mrun, tmax);
    const float esc = __expf(mrun - newm);

    float lsum = 0.0f;
    v16h bp;
    #pragma unroll
    for (int i = 0; i < 16; ++i) {
      float e = __expf(p[i] - newm);
      lsum += e;
      bp[i] = (_Float16)e;   // exactly B-frag element order for 32-key chunk
    }
    lsum += __shfl_xor(lsum, 16, 32);
    lrun = lrun * esc + lsum;
    mrun = newm;

    #pragma unroll
    for (int f = 0; f < 4; ++f)
      #pragma unroll
      for (int v = 0; v < 8; ++v) o[f][v] *= esc;

    #pragma unroll
    for (int f = 0; f < 4; ++f) {
      v16h va = ldfrag(&Vs[buf][(f * 16 + lq) * LDV_PAD + 8 * half]);
      o[f] = wmma_f16(va, bp, o[f]);
    }
  }

  const float inv = 1.0f / lrun;
  _Float16* crow =
      ctx + (size_t)(b * SEQ + qt * 16 + lq) * D_MODEL + h * DK + 8 * half;
  #pragma unroll
  for (int f = 0; f < 4; ++f) {
    h8 pk;
    #pragma unroll
    for (int v = 0; v < 8; ++v) pk[v] = (_Float16)(o[f][v] * inv);
    *(h8*)(crow + f * 16) = pk;
  }
}

// ---------------------------------------------------------------------------
extern "C" void kernel_launch(void* const* d_in, const int* in_sizes, int n_in,
                              void* d_out, int out_size, void* d_ws, size_t ws_size,
                              hipStream_t stream)
{
  const float* x     = (const float*)d_in[0];
  const int*   mask  = (const int*)d_in[1];
  const float* wq    = (const float*)d_in[2];
  const float* biasq = (const float*)d_in[3];
  const float* wk    = (const float*)d_in[4];
  const float* biask = (const float*)d_in[5];
  const float* wv    = (const float*)d_in[6];
  const float* biasv = (const float*)d_in[7];
  const float* wo    = (const float*)d_in[8];
  const float* biaso = (const float*)d_in[9];
  const float* w1    = (const float*)d_in[10];
  const float* bias1 = (const float*)d_in[11];
  const float* w2    = (const float*)d_in[12];
  const float* bias2 = (const float*)d_in[13];
  const float* ln1a  = (const float*)d_in[14];
  const float* ln1b  = (const float*)d_in[15];
  const float* ln2a  = (const float*)d_in[16];
  const float* ln2b  = (const float*)d_in[17];
  float* out = (float*)d_out;
  (void)in_sizes; (void)n_in; (void)out_size; (void)ws_size;

  char* p = (char*)d_ws;
  auto carve = [&](size_t bytes) -> void* {
    void* r = (void*)p;
    p += (bytes + 255) & ~((size_t)255);
    return r;
  };
  _Float16* h16   = (_Float16*)carve((size_t)TOKENS * D_MODEL * 2);
  _Float16* q16   = (_Float16*)carve((size_t)TOKENS * D_MODEL * 2);
  _Float16* k16   = (_Float16*)carve((size_t)TOKENS * D_MODEL * 2);
  _Float16* vt16  = (_Float16*)carve((size_t)TOKENS * D_MODEL * 2);
  _Float16* ctx16 = (_Float16*)carve((size_t)TOKENS * D_MODEL * 2);
  float*    x2    = (float*)   carve((size_t)TOKENS * D_MODEL * 4);
  _Float16* h2    = (_Float16*)carve((size_t)TOKENS * D_MODEL * 2);
  _Float16* ffn16 = (_Float16*)carve((size_t)TOKENS * D_FF * 2);
  _Float16* wtq   = (_Float16*)carve((size_t)D_MODEL * D_MODEL * 2);
  _Float16* wtk   = (_Float16*)carve((size_t)D_MODEL * D_MODEL * 2);
  _Float16* wtv   = (_Float16*)carve((size_t)D_MODEL * D_MODEL * 2);
  _Float16* wto   = (_Float16*)carve((size_t)D_MODEL * D_MODEL * 2);
  _Float16* wt1   = (_Float16*)carve((size_t)D_MODEL * D_FF * 2);
  _Float16* wt2   = (_Float16*)carve((size_t)D_MODEL * D_FF * 2);

  // Weight prep: [K,N] fp32 -> [N,K] f16
  transpose_cast_kernel<<<(D_MODEL*D_MODEL)/256, 256, 0, stream>>>(wq, wtq, D_MODEL, D_MODEL);
  transpose_cast_kernel<<<(D_MODEL*D_MODEL)/256, 256, 0, stream>>>(wk, wtk, D_MODEL, D_MODEL);
  transpose_cast_kernel<<<(D_MODEL*D_MODEL)/256, 256, 0, stream>>>(wv, wtv, D_MODEL, D_MODEL);
  transpose_cast_kernel<<<(D_MODEL*D_MODEL)/256, 256, 0, stream>>>(wo, wto, D_MODEL, D_MODEL);
  transpose_cast_kernel<<<(D_MODEL*D_FF)/256,    256, 0, stream>>>(w1, wt1, D_MODEL, D_FF);
  transpose_cast_kernel<<<(D_MODEL*D_FF)/256,    256, 0, stream>>>(w2, wt2, D_FF, D_MODEL);

  // h = LN1(x)
  ln_kernel<<<TOKENS, 256, 0, stream>>>(x, h16, ln1a, ln1b);

  // Q, K (f16), V (f16, transposed to [B,H,dk,S])
  dim3 g768(D_MODEL/128, TOKENS/128);
  gemm_kernel<false,false,true, false,false><<<g768, 256, 0, stream>>>(
      h16, wtq, biasq, nullptr, nullptr, q16, nullptr, TOKENS, D_MODEL, D_MODEL);
  gemm_kernel<false,false,true, false,false><<<g768, 256, 0, stream>>>(
      h16, wtk, biask, nullptr, nullptr, k16, nullptr, TOKENS, D_MODEL, D_MODEL);
  gemm_kernel<false,false,false,false,true ><<<g768, 256, 0, stream>>>(
      h16, wtv, biasv, nullptr, nullptr, nullptr, vt16, TOKENS, D_MODEL, D_MODEL);

  // Flash attention -> ctx (f16)
  attn_kernel<<<BATCH * N_HEADS * 32, 128, 0, stream>>>(
      q16, k16, vt16, mask, ctx16);

  // x2 = x + ctx @ Wo + bo  (fp32)
  gemm_kernel<false,true, false,true, false><<<g768, 256, 0, stream>>>(
      ctx16, wto, biaso, x, x2, nullptr, nullptr, TOKENS, D_MODEL, D_MODEL);

  // h2 = LN2(x2)
  ln_kernel<<<TOKENS, 256, 0, stream>>>(x2, h2, ln2a, ln2b);

  // ffn = relu(h2 @ W1 + b1)  (f16)
  dim3 gff(D_FF/128, TOKENS/128);
  gemm_kernel<true, false,true, false,false><<<gff, 256, 0, stream>>>(
      h2, wt1, bias1, nullptr, nullptr, ffn16, nullptr, TOKENS, D_FF, D_MODEL);

  // out = x2 + ffn @ W2 + b2  (fp32, straight to d_out)
  gemm_kernel<false,true, false,true, false><<<g768, 256, 0, stream>>>(
      ffn16, wt2, bias2, x2, out, nullptr, nullptr, TOKENS, D_MODEL, D_FF);
}
